// PlainSignLinear_84902913508067
// MI455X (gfx1250) — matrix operand order repaired and checked by reading, not compiled, not run on hardware
//
#include <hip/hip_runtime.h>
#include <stdint.h>

// y[m, n] = (sum_k x[m, k] * sign[n, k]) * scale[n] + bias[n]
// M = 8192, N = 4096, K = 4096, fp32 in memory.
//
// Pass 1 (memory-bound, once): x -> bf16 hi/lo planes, sign -> bf16, into d_ws.
//   ws layout: A_hi [8192x4096] bf16 @ 0        (64 MiB)
//              A_lo [8192x4096] bf16 @ 64 MiB   (64 MiB)
//              Bsgn [4096x4096] bf16 @ 128 MiB  (32 MiB)   => needs ws >= 160 MiB
//   Working set now fits in the 192 MB L2 -> all tile re-reads are L2 hits.
// Pass 2: WMMA GEMM. Hot loop has zero conversion VALU; global->LDS staging uses
//   CDNA5 async copies (global_load_async_to_lds_b128, ASYNCcnt-tracked).

#define M_DIM 8192
#define N_DIM 4096
#define K_DIM 4096

#define TILE_M 128
#define TILE_N 128
#define TILE_K 32
#define LDSK   40                          // padded LDS row stride in bf16 elems (80 B)
#define REGION_ELEMS (TILE_M * LDSK)       // 5120 bf16 per region
#define REGION_BYTES (REGION_ELEMS * 2)    // 10240 B
#define A_HI_E 0
#define A_LO_E REGION_ELEMS
#define B_E    (2 * REGION_ELEMS)
#define BUF_ELEMS (3 * REGION_ELEMS)       // 15360
#define BUF_BYTES (BUF_ELEMS * 2)          // 30720

typedef __bf16 v16bf __attribute__((ext_vector_type(16)));
typedef __bf16 v8bf  __attribute__((ext_vector_type(8)));
typedef float  v8f   __attribute__((ext_vector_type(8)));
typedef unsigned short us4 __attribute__((ext_vector_type(4)));

#define CAT16(a, b) __builtin_shufflevector(a, b, 0,1,2,3,4,5,6,7,8,9,10,11,12,13,14,15)

__device__ __forceinline__ unsigned short f2bf(float f) {
    unsigned int u = __float_as_uint(f);
    u += 0x7FFFu + ((u >> 16) & 1u);       // RNE (finite inputs)
    return (unsigned short)(u >> 16);
}
__device__ __forceinline__ float bf2f(unsigned short h) {
    return __uint_as_float(((unsigned int)h) << 16);
}

// ---------------- pass 1: conversions (run once, memory-bound) ----------------

__global__ __launch_bounds__(256)
void convert_x_kernel(const float* __restrict__ x,
                      unsigned short* __restrict__ ahi,
                      unsigned short* __restrict__ alo) {
    size_t i = (size_t)blockIdx.x * 256 + threadIdx.x;     // float4 index
    float4 v = ((const float4*)x)[i];
    float a[4] = {v.x, v.y, v.z, v.w};
    us4 hi, lo;
#pragma unroll
    for (int j = 0; j < 4; ++j) {
        unsigned short h = f2bf(a[j]);
        hi[j] = h;
        lo[j] = f2bf(a[j] - bf2f(h));
    }
    ((us4*)ahi)[i] = hi;
    ((us4*)alo)[i] = lo;
}

__global__ __launch_bounds__(256)
void convert_sign_kernel(const float* __restrict__ sg,
                         unsigned short* __restrict__ bsg) {
    size_t i = (size_t)blockIdx.x * 256 + threadIdx.x;
    float4 v = ((const float4*)sg)[i];
    us4 b;
    b[0] = f2bf(v.x); b[1] = f2bf(v.y); b[2] = f2bf(v.z); b[3] = f2bf(v.w);
    ((us4*)bsg)[i] = b;
}

// ---------------- pass 2: WMMA GEMM with async global->LDS staging ----------------

__global__ __launch_bounds__(256)
void sign_linear_wmma_kernel(const unsigned short* __restrict__ ahi,
                             const unsigned short* __restrict__ alo,
                             const unsigned short* __restrict__ bsg,
                             const float* __restrict__ scale,
                             const float* __restrict__ bias,
                             float* __restrict__ out) {
    __shared__ unsigned short lds[2 * BUF_ELEMS];   // 61440 bytes

    const int tid  = threadIdx.x;
    const int lane = tid & 31;
    const int wave = tid >> 5;
    const int wm   = wave >> 2;     // 0..1 : 64-row slab
    const int wn   = wave & 3;      // 0..3 : 32-col slab
    const int m0   = blockIdx.y * TILE_M;
    const int n0   = blockIdx.x * TILE_N;

    // LDS byte address of our shared array (flat addr truncated to 32b == LDS offset)
    const unsigned lds_base = (unsigned)(uintptr_t)(&lds[0]);

    // Each thread owns 6 16-byte chunks per K-step (1536 chunks = 24 KB / block):
    // chunks [0,512) -> A_hi, [512,1024) -> A_lo, [1024,1536) -> B. 4 chunks per row.
    unsigned lds_off[6];
    unsigned long long gbase[6];
#pragma unroll
    for (int i = 0; i < 6; ++i) {
        int c      = tid + i * 256;
        int region = c >> 9;
        int cc     = c & 511;
        int row    = cc >> 2;
        int c4     = cc & 3;
        lds_off[i] = lds_base + (unsigned)(region * REGION_BYTES + row * (LDSK * 2) + c4 * 16);
        const unsigned short* gp;
        if (region == 0)      gp = ahi + (size_t)(m0 + row) * K_DIM;
        else if (region == 1) gp = alo + (size_t)(m0 + row) * K_DIM;
        else                  gp = bsg + (size_t)(n0 + row) * K_DIM;
        gbase[i] = (unsigned long long)(uintptr_t)gp + (unsigned)(c4 * 16);
    }

    auto issue_async = [&](int kt, int buf) {
        const unsigned long long koff = (unsigned long long)kt * (TILE_K * 2);
        const unsigned lbuf = (unsigned)(buf * BUF_BYTES);
#pragma unroll
        for (int i = 0; i < 6; ++i) {
            unsigned la = lds_off[i] + lbuf;
            unsigned long long ga = gbase[i] + koff;
            asm volatile("global_load_async_to_lds_b128 %0, %1, off"
                         :: "v"(la), "v"(ga) : "memory");
        }
    };

    v8f acc[4][2];
#pragma unroll
    for (int tm = 0; tm < 4; ++tm)
#pragma unroll
        for (int tn = 0; tn < 2; ++tn)
            acc[tm][tn] = (v8f){0.f, 0.f, 0.f, 0.f, 0.f, 0.f, 0.f, 0.f};

    issue_async(0, 0);

    const int TSTEPS = K_DIM / TILE_K;   // 128
    for (int kt = 0; kt < TSTEPS; ++kt) {
        const int buf = kt & 1;

        if (kt + 1 < TSTEPS) {
            issue_async(kt + 1, buf ^ 1);                      // overlap with compute
            asm volatile("s_wait_asynccnt 0x6" ::: "memory");  // oldest batch (this step) landed
        } else {
            asm volatile("s_wait_asynccnt 0x0" ::: "memory");
        }
        __syncthreads();   // all waves' async data for `buf` visible

        // ---- fragments from LDS + WMMA ----
        {
            const unsigned short* base = lds + buf * BUF_ELEMS;
            // A (16x32 bf16): lane<16 -> K {0..7,16..23}; lane>=16 -> K {8..15,24..31}
            const int mrow = lane & 15;
            const int kbA  = (lane >> 4) * 8;
            v16bf ah[4], al[4];
#pragma unroll
            for (int tm = 0; tm < 4; ++tm) {
                int row = wm * 64 + tm * 16 + mrow;
                const v8bf* h0 = (const v8bf*)(base + A_HI_E + row * LDSK + kbA);
                const v8bf* h1 = (const v8bf*)(base + A_HI_E + row * LDSK + kbA + 16);
                ah[tm] = CAT16(h0[0], h1[0]);
                const v8bf* l0 = (const v8bf*)(base + A_LO_E + row * LDSK + kbA);
                const v8bf* l1 = (const v8bf*)(base + A_LO_E + row * LDSK + kbA + 16);
                al[tm] = CAT16(l0[0], l1[0]);
            }
            // B (32x16 bf16): lane<16 -> col N=lane, K {0..7,16..23}; lane>=16 -> K {8..15,24..31}
            const int ncol = lane & 15;
            const int kbB  = (lane >> 4) * 8;
            v16bf bfr[2];
#pragma unroll
            for (int tn = 0; tn < 2; ++tn) {
                int row = wn * 32 + tn * 16 + ncol;            // LDS holds sign tile as [n][k]
                const v8bf* q = (const v8bf*)(base + B_E + row * LDSK + kbB);
                bfr[tn] = CAT16(q[0], q[2]);
            }
#pragma unroll
            for (int tm = 0; tm < 4; ++tm)
#pragma unroll
                for (int tn = 0; tn < 2; ++tn) {
                    acc[tm][tn] = __builtin_amdgcn_wmma_f32_16x16x32_bf16(
                        false, ah[tm], false, bfr[tn], (short)0, acc[tm][tn], false, false);
                    acc[tm][tn] = __builtin_amdgcn_wmma_f32_16x16x32_bf16(
                        false, al[tm], false, bfr[tn], (short)0, acc[tm][tn], false, false);
                }
        }

        __syncthreads();   // safe to overwrite `buf` with the kt+2 issue next iteration
    }

    // ---- epilogue: y = acc * scale[n] + bias[n] ----
    // C layout: VGPR r, lanes 0-15 -> (M=r, N=lane); lanes 16-31 -> (M=8+r, N=lane-16)
    {
        const int ncol  = lane & 15;
        const int mhalf = lane >> 4;
#pragma unroll
        for (int tn = 0; tn < 2; ++tn) {
            int n   = n0 + wn * 32 + tn * 16 + ncol;
            float s = scale[n];
            float b = bias[n];
#pragma unroll
            for (int tm = 0; tm < 4; ++tm) {
                int mbase = m0 + wm * 64 + tm * 16 + mhalf * 8;
#pragma unroll
                for (int r = 0; r < 8; ++r) {
                    out[(size_t)(mbase + r) * N_DIM + n] = acc[tm][tn][r] * s + b;
                }
            }
        }
    }
}

extern "C" void kernel_launch(void* const* d_in, const int* in_sizes, int n_in,
                              void* d_out, int out_size, void* d_ws, size_t ws_size,
                              hipStream_t stream) {
    (void)in_sizes; (void)n_in; (void)out_size; (void)ws_size;
    const float* x     = (const float*)d_in[0];
    const float* sign  = (const float*)d_in[1];
    const float* scale = (const float*)d_in[2];
    const float* bias  = (const float*)d_in[3];
    float* out = (float*)d_out;

    // ws partitions (bf16 planes); requires ws_size >= 160 MiB
    unsigned short* ahi = (unsigned short*)d_ws;
    unsigned short* alo = ahi + (size_t)M_DIM * K_DIM;          // +64 MiB
    unsigned short* bsg = alo + (size_t)M_DIM * K_DIM;          // +128 MiB (32 MiB used)

    convert_x_kernel<<<(M_DIM * (size_t)K_DIM / 4) / 256, 256, 0, stream>>>(x, ahi, alo);
    convert_sign_kernel<<<(N_DIM * (size_t)K_DIM / 4) / 256, 256, 0, stream>>>(sign, bsg);

    dim3 grid(N_DIM / TILE_N, M_DIM / TILE_M);   // 32 x 64
    sign_linear_wmma_kernel<<<grid, 256, 0, stream>>>(ahi, alo, bsg, scale, bias, out);
}
